// AttentionBlock_50577534878336
// MI455X (gfx1250) — matrix operand order repaired and checked by reading, not compiled, not run on hardware
//
#include <hip/hip_runtime.h>
#include <hip/hip_bf16.h>

typedef __attribute__((ext_vector_type(16))) _Float16 v16h;
typedef __attribute__((ext_vector_type(8)))  _Float16 v8h;
typedef __attribute__((ext_vector_type(8)))  float    v8f;

#define WMMA_F16(a, b, c) \
  __builtin_amdgcn_wmma_f32_16x16x32_f16(false, (a), false, (b), (short)0, (c), false, false)

constexpr int  B_  = 4;
constexpr int  C_  = 512;
constexpr int  T_  = 2048;
constexpr int  H_  = 8;
constexpr int  CH  = 64;      // channels per head
constexpr int  G_  = 32;      // groupnorm groups
constexpr int  O3  = 3 * C_;  // 1536
constexpr float SCALE = 0.35355339059327373f;  // 64^-0.25

// async global->LDS copy, 16B per lane, tracked by ASYNCcnt
__device__ __forceinline__ void async_b128(const void* g, void* lds_ptr) {
  unsigned l = (unsigned)(uintptr_t)lds_ptr;  // low 32 bits of generic LDS ptr = LDS byte addr
  asm volatile("global_load_async_to_lds_b128 %0, %1, off" :: "v"(l), "v"(g) : "memory");
}

// ---------------------------------------------------------------- weights cvt
__global__ void k_cvt(const float* __restrict__ qkv_w, const float* __restrict__ proj_w,
                      _Float16* __restrict__ wq, _Float16* __restrict__ wp) {
  int i = blockIdx.x * 256 + threadIdx.x;
  if (i < O3 * C_) wq[i] = (_Float16)qkv_w[i];
  if (i < C_ * C_) wp[i] = (_Float16)proj_w[i];
}

// ---------------------------------------------------------------- group norm
__global__ void k_gn(const float* __restrict__ x, const float* __restrict__ gw,
                     const float* __restrict__ gb, _Float16* __restrict__ xnT) {
  const int CPG = C_ / G_;  // 16
  int b = blockIdx.x / G_;
  int g = blockIdx.x % G_;
  const float* xp = x + ((size_t)b * C_ + (size_t)g * CPG) * T_;

  float s = 0.f, ss = 0.f;
  for (int i = threadIdx.x; i < CPG * T_; i += 256) {
    float v = xp[i];
    s += v; ss += v * v;
  }
  __shared__ float rs[8], rss[8];
  for (int msk = 16; msk >= 1; msk >>= 1) {
    s  += __shfl_xor(s,  msk, 32);
    ss += __shfl_xor(ss, msk, 32);
  }
  int wave = threadIdx.x >> 5, lane = threadIdx.x & 31;
  if (lane == 0) { rs[wave] = s; rss[wave] = ss; }
  __syncthreads();
  if (wave == 0) {
    s  = (lane < 8) ? rs[lane]  : 0.f;
    ss = (lane < 8) ? rss[lane] : 0.f;
    for (int msk = 4; msk >= 1; msk >>= 1) {
      s  += __shfl_xor(s,  msk, 32);
      ss += __shfl_xor(ss, msk, 32);
    }
    if (lane == 0) { rs[0] = s; rss[0] = ss; }
  }
  __syncthreads();
  float N    = (float)(CPG * T_);
  float mu   = rs[0] / N;
  float var  = rss[0] / N - mu * mu;
  float rinv = rsqrtf(var + 1e-5f);
  for (int i = threadIdx.x; i < CPG * T_; i += 256) {
    int cl = i / T_, t = i % T_;
    int c  = g * CPG + cl;
    float v = (xp[i] - mu) * rinv * gw[c] + gb[c];
    xnT[((size_t)b * T_ + t) * C_ + c] = (_Float16)v;
  }
}

// ---------------------------------------------------------------- QKV GEMM
__global__ __launch_bounds__(256) void k_qkv(
    const _Float16* __restrict__ wq, const _Float16* __restrict__ xnT,
    const float* __restrict__ qkv_b, const float* __restrict__ pos,
    _Float16* __restrict__ qT, _Float16* __restrict__ kT, _Float16* __restrict__ vm) {
  int wave = threadIdx.x >> 5, lane = threadIdx.x & 31;
  int ln = lane & 15, hi = lane >> 4;
  int tile = blockIdx.x * 8 + wave;
  const int OT = O3 / 16, TT = T_ / 16;  // 96, 128
  int bidx = tile / (OT * TT);
  int rem  = tile % (OT * TT);
  int obase = (rem / TT) * 16;
  int tbase = (rem % TT) * 16;

  const _Float16* arow = wq  + (size_t)(obase + ln) * C_;
  const _Float16* brow = xnT + ((size_t)bidx * T_ + (tbase + ln)) * C_;

  v8f acc = {};
  for (int k0 = 0; k0 < C_; k0 += 32) {
    union { v16h v; v8h h[2]; } ua;
    ua.h[0] = *(const v8h*)(arow + k0 + (hi ? 16 : 0));
    ua.h[1] = *(const v8h*)(arow + k0 + 8 + (hi ? 16 : 0));
    v16h bm = *(const v16h*)(brow + k0 + (hi ? 16 : 0));
    acc = WMMA_F16(ua.v, bm, acc);
  }

  // reshape (B, 3C, T) -> (B*H, 192, T) then split: head=o/192, which=(o%192)/64
  int head  = obase / 192;
  int which = (obase % 192) / 64;
  int c0    = obase % 64;
  int bh    = bidx * H_ + head;
  int t     = tbase + ln;

  if (which == 2) {  // V: natural [bh][c][T] layout for PV B-tiles
    for (int r = 0; r < 8; r++) {
      int o = obase + r + 8 * hi;
      int c = c0 + r + 8 * hi;
      float v = acc[r] + qkv_b[o] + pos[((size_t)bh * CH + c) * T_ + t];
      vm[((size_t)bh * CH + c) * T_ + t] = (_Float16)v;
    }
  } else {           // Q/K: transposed [bh][T][64], pre-scaled
    v8h pk;
    for (int r = 0; r < 8; r++) {
      int o = obase + r + 8 * hi;
      int c = c0 + r + 8 * hi;
      float v = (acc[r] + qkv_b[o] + pos[((size_t)bh * CH + c) * T_ + t]) * SCALE;
      pk[r] = (_Float16)v;
    }
    _Float16* dst = (which == 0 ? qT : kT) + ((size_t)bh * T_ + t) * CH + c0 + 8 * hi;
    *(v8h*)dst = pk;
  }
}

// ---------------------------------------------------------------- attention
// One block per (bh, 128-query slab); 8 waves x 16 queries. K/V strips are
// staged into LDS once per block via async global->LDS copies, double-buffered.
constexpr int KROW = 72;  // padded row stride (halves) for K stage: 144B -> conflict-free
constexpr int VROW = 40;  // padded row stride for V stage: 80B
constexpr int PROW = 40;  // padded row stride for P bounce

__global__ __launch_bounds__(256) void k_attn(
    const _Float16* __restrict__ qT, const _Float16* __restrict__ kT,
    const _Float16* __restrict__ vm, const unsigned char* __restrict__ mask,
    _Float16* __restrict__ hT) {
  __shared__ __align__(16) _Float16 kbuf[2][32 * KROW];
  __shared__ __align__(16) _Float16 vbuf[2][64 * VROW];
  __shared__ __align__(16) _Float16 plds_all[8 * 16 * PROW];

  int tid  = threadIdx.x;
  int wave = tid >> 5, lane = tid & 31;
  int ln = lane & 15, hi = lane >> 4;
  int bh    = blockIdx.x >> 4;
  int qbase = ((blockIdx.x & 15) * 8 + wave) * 16;
  int b = bh / H_, head = bh % H_;
  int mb = bh % B_;  // torch mask.repeat(H,1,1) tiles head-major: mask[bh % B]

  const _Float16* kbase = kT + (size_t)bh * T_ * CH;
  const _Float16* vbase = vm + (size_t)bh * CH * T_;

  // Q as A-matrix, two K=32 channel chunks (loaded once)
  const _Float16* qrow = qT + ((size_t)bh * T_ + qbase + ln) * CH;
  union { v16h v; v8h h[2]; } a0, a1;
  a0.h[0] = *(const v8h*)(qrow + 0  + (hi ? 16 : 0));
  a0.h[1] = *(const v8h*)(qrow + 8  + (hi ? 16 : 0));
  a1.h[0] = *(const v8h*)(qrow + 32 + (hi ? 16 : 0));
  a1.h[1] = *(const v8h*)(qrow + 40 + (hi ? 16 : 0));

  v8f acc0 = {}, acc1 = {}, acc2 = {}, acc3 = {};
  float m[8], l[8];
  for (int r = 0; r < 8; r++) { m[r] = -3.0e38f; l[r] = 0.f; }

  _Float16* plds = plds_all + wave * 16 * PROW;

  // staging coordinates for this thread (one b128 per buffer per strip)
  int krow = tid >> 3, kseg = (tid & 7) * 8;   // K: 32 rows x 64 halves
  int vrow = tid >> 2, vseg = (tid & 3) * 8;   // V: 64 rows x 32 halves

  const int NSTRIP = T_ / 32;  // 64

  // prologue: stage strip 0 into buffer 0
  async_b128(kbase + (size_t)(0 + krow) * CH + kseg, &kbuf[0][krow * KROW + kseg]);
  async_b128(vbase + (size_t)vrow * T_ + 0 + vseg,   &vbuf[0][vrow * VROW + vseg]);

  for (int i = 0; i < NSTRIP; i++) {
    int s0 = i * 32;
    int p  = i & 1;
    if (i + 1 < NSTRIP) {
      int s1 = s0 + 32, p1 = p ^ 1;
      async_b128(kbase + (size_t)(s1 + krow) * CH + kseg, &kbuf[p1][krow * KROW + kseg]);
      async_b128(vbase + (size_t)vrow * T_ + s1 + vseg,   &vbuf[p1][vrow * VROW + vseg]);
      asm volatile("s_wait_asynccnt 0x2" ::: "memory");  // strip i's copies done
    } else {
      asm volatile("s_wait_asynccnt 0x0" ::: "memory");
    }
    __syncthreads();  // all waves' staging of buffer p visible

    const _Float16* kb = kbuf[p];
    const _Float16* vb = vbuf[p];

    // ----- S = Q^T K for 16 queries x 32 keys (4 WMMAs), B tiles from LDS
    union { v16h v; v8h h[2]; } bk00, bk01, bk10, bk11;
    {
      const _Float16* r0 = kb + (size_t)ln * KROW + hi * 16;
      const _Float16* r1 = kb + (size_t)(16 + ln) * KROW + hi * 16;
      bk00.h[0] = *(const v8h*)(r0);      bk00.h[1] = *(const v8h*)(r0 + 8);
      bk01.h[0] = *(const v8h*)(r0 + 32); bk01.h[1] = *(const v8h*)(r0 + 40);
      bk10.h[0] = *(const v8h*)(r1);      bk10.h[1] = *(const v8h*)(r1 + 8);
      bk11.h[0] = *(const v8h*)(r1 + 32); bk11.h[1] = *(const v8h*)(r1 + 40);
    }
    v8f s0v = {}, s1v = {};
    s0v = WMMA_F16(a0.v, bk00.v, s0v);
    s0v = WMMA_F16(a1.v, bk01.v, s0v);
    s1v = WMMA_F16(a0.v, bk10.v, s1v);
    s1v = WMMA_F16(a1.v, bk11.v, s1v);

    // ----- key mask (column-wise)
    unsigned char mk0 = mask[(size_t)mb * T_ + s0 + ln];
    unsigned char mk1 = mask[(size_t)mb * T_ + s0 + 16 + ln];
    if (mk0) for (int r = 0; r < 8; r++) s0v[r] = -1e9f;
    if (mk1) for (int r = 0; r < 8; r++) s1v[r] = -1e9f;

    // ----- online softmax (row stats live per half-wave: xor 1/2/4/8)
    float scl[8];
    for (int r = 0; r < 8; r++) {
      float mx = fmaxf(s0v[r], s1v[r]);
      mx = fmaxf(mx, __shfl_xor(mx, 1, 32));
      mx = fmaxf(mx, __shfl_xor(mx, 2, 32));
      mx = fmaxf(mx, __shfl_xor(mx, 4, 32));
      mx = fmaxf(mx, __shfl_xor(mx, 8, 32));
      float mn = fmaxf(m[r], mx);
      scl[r] = __expf(m[r] - mn);
      m[r] = mn;
      float p0 = __expf(s0v[r] - mn);
      float p1 = __expf(s1v[r] - mn);
      s0v[r] = p0; s1v[r] = p1;
      float sum = p0 + p1;
      sum += __shfl_xor(sum, 1, 32);
      sum += __shfl_xor(sum, 2, 32);
      sum += __shfl_xor(sum, 4, 32);
      sum += __shfl_xor(sum, 8, 32);
      l[r] = l[r] * scl[r] + sum;
    }
    for (int r = 0; r < 8; r++) {
      acc0[r] *= scl[r]; acc1[r] *= scl[r];
      acc2[r] *= scl[r]; acc3[r] *= scl[r];
    }

    // ----- P: D-layout f32 -> (LDS bounce) -> A-layout f16
    for (int r = 0; r < 8; r++) {
      int row = r + 8 * hi;
      plds[row * PROW + ln]      = (_Float16)s0v[r];
      plds[row * PROW + 16 + ln] = (_Float16)s1v[r];
    }
    asm volatile("s_wait_dscnt 0" ::: "memory");  // wave-synchronous cross-lane RAW
    union { v16h v; v8h h[2]; } pa;
    pa.h[0] = *(const v8h*)(plds + ln * PROW + (hi ? 16 : 0));
    pa.h[1] = *(const v8h*)(plds + ln * PROW + 8 + (hi ? 16 : 0));

    // ----- O += P V  (4 WMMAs over 4 channel tiles), B tiles from LDS
    union { v16h v; v8h h[2]; } bv0, bv1, bv2, bv3;
    bv0.h[0] = *(const v8h*)(vb + (size_t)(ln)      * VROW + hi * 16);
    bv0.h[1] = *(const v8h*)(vb + (size_t)(ln)      * VROW + hi * 16 + 8);
    bv1.h[0] = *(const v8h*)(vb + (size_t)(16 + ln) * VROW + hi * 16);
    bv1.h[1] = *(const v8h*)(vb + (size_t)(16 + ln) * VROW + hi * 16 + 8);
    bv2.h[0] = *(const v8h*)(vb + (size_t)(32 + ln) * VROW + hi * 16);
    bv2.h[1] = *(const v8h*)(vb + (size_t)(32 + ln) * VROW + hi * 16 + 8);
    bv3.h[0] = *(const v8h*)(vb + (size_t)(48 + ln) * VROW + hi * 16);
    bv3.h[1] = *(const v8h*)(vb + (size_t)(48 + ln) * VROW + hi * 16 + 8);
    acc0 = WMMA_F16(pa.v, bv0.v, acc0);
    acc1 = WMMA_F16(pa.v, bv1.v, acc1);
    acc2 = WMMA_F16(pa.v, bv2.v, acc2);
    acc3 = WMMA_F16(pa.v, bv3.v, acc3);

    __syncthreads();  // all waves done reading buffer p before it is restaged
  }

  // ----- normalize + store hT[b][t][C] for proj GEMM
  for (int r = 0; r < 8; r++) {
    int t = qbase + r + 8 * hi;
    float inv = 1.f / l[r];
    size_t base = ((size_t)b * T_ + t) * C_ + (size_t)head * CH + ln;
    hT[base +  0] = (_Float16)(acc0[r] * inv);
    hT[base + 16] = (_Float16)(acc1[r] * inv);
    hT[base + 32] = (_Float16)(acc2[r] * inv);
    hT[base + 48] = (_Float16)(acc3[r] * inv);
  }
}

// ---------------------------------------------------------------- proj + residual
__global__ __launch_bounds__(256) void k_proj(
    const _Float16* __restrict__ wp, const _Float16* __restrict__ hT,
    const float* __restrict__ proj_b, const float* __restrict__ x,
    float* __restrict__ out) {
  int wave = threadIdx.x >> 5, lane = threadIdx.x & 31;
  int ln = lane & 15, hi = lane >> 4;
  int tile = blockIdx.x * 8 + wave;
  const int OT = C_ / 16, TT = T_ / 16;  // 32, 128
  int bidx = tile / (OT * TT);
  int rem  = tile % (OT * TT);
  int obase = (rem / TT) * 16;
  int tbase = (rem % TT) * 16;

  const _Float16* arow = wp + (size_t)(obase + ln) * C_;
  const _Float16* brow = hT + ((size_t)bidx * T_ + tbase + ln) * C_;

  v8f acc = {};
  for (int k0 = 0; k0 < C_; k0 += 32) {
    union { v16h v; v8h h[2]; } ua;
    ua.h[0] = *(const v8h*)(arow + k0 + (hi ? 16 : 0));
    ua.h[1] = *(const v8h*)(arow + k0 + 8 + (hi ? 16 : 0));
    v16h bm = *(const v16h*)(brow + k0 + (hi ? 16 : 0));
    acc = WMMA_F16(ua.v, bm, acc);
  }
  int t = tbase + ln;
  for (int r = 0; r < 8; r++) {
    int o = obase + r + 8 * hi;
    size_t idx = ((size_t)bidx * C_ + o) * T_ + t;
    out[idx] = x[idx] + acc[r] + proj_b[o];
  }
}

// ---------------------------------------------------------------- launch
extern "C" void kernel_launch(void* const* d_in, const int* in_sizes, int n_in,
                              void* d_out, int out_size, void* d_ws, size_t ws_size,
                              hipStream_t stream) {
  const float*         x      = (const float*)d_in[0];
  const float*         pos    = (const float*)d_in[1];
  const unsigned char* mask   = (const unsigned char*)d_in[2];  // bool
  const float*         gn_w   = (const float*)d_in[3];
  const float*         gn_b   = (const float*)d_in[4];
  const float*         qkv_w  = (const float*)d_in[5];
  const float*         qkv_b  = (const float*)d_in[6];
  const float*         proj_w = (const float*)d_in[7];
  const float*         proj_b = (const float*)d_in[8];
  float*               out    = (float*)d_out;

  char* ws = (char*)d_ws;
  _Float16* xnT  = (_Float16*)(ws);                          // B*T*C   f16 = 8 MB
  _Float16* qT   = (_Float16*)(ws + ((size_t)8  << 20));     // bh*T*64 f16 = 8 MB
  _Float16* kT   = (_Float16*)(ws + ((size_t)16 << 20));     // 8 MB
  _Float16* vm   = (_Float16*)(ws + ((size_t)24 << 20));     // 8 MB
  _Float16* hT   = (_Float16*)(ws + ((size_t)32 << 20));     // 8 MB
  _Float16* wq16 = (_Float16*)(ws + ((size_t)40 << 20));     // 1.5 MB
  _Float16* wp16 = (_Float16*)(ws + ((size_t)40 << 20) + (size_t)O3 * C_ * 2);

  k_cvt <<<(O3 * C_) / 256, 256, 0, stream>>>(qkv_w, proj_w, wq16, wp16);
  k_gn  <<<B_ * G_,          256, 0, stream>>>(x, gn_w, gn_b, xnT);
  k_qkv <<<(B_ * 96 * 128) / 8, 256, 0, stream>>>(wq16, xnT, qkv_b, pos, qT, kT, vm);
  k_attn<<<(B_ * H_) * (T_ / 16 / 8), 256, 0, stream>>>(qT, kT, vm, mask, hT);
  k_proj<<<(B_ * 32 * 128) / 8, 256, 0, stream>>>(wp16, hT, proj_b, x, out);
}
